// SimpleTemporalGNN_59889023976185
// MI455X (gfx1250) — matrix operand order repaired and checked by reading, not compiled, not run on hardware
//
#include <hip/hip_runtime.h>
#include <hip/hip_bf16.h>
#include <math.h>

typedef __attribute__((ext_vector_type(16))) __bf16 v16bf;
typedef __attribute__((ext_vector_type(8)))  float  v8f;

#define NNODES 50000
#define TSTEPS 8
#define NEDGES 1600000
#define HDIM   128
#define NH     (NNODES * HDIM)   // 6,400,000

// ---------------------------------------------------------------------------
// Weight prep: W1,W2 -> bf16 transposed (col-major: row j holds K contiguous),
// Wih,Whh -> bf16 (already [4H][H] row-major = K contiguous per output col),
// bsum = bih + bhh.
// ---------------------------------------------------------------------------
__global__ void prep_weights(const float* __restrict__ W1, const float* __restrict__ W2,
                             const float* __restrict__ Wih, const float* __restrict__ Whh,
                             const float* __restrict__ bih, const float* __restrict__ bhh,
                             __bf16* __restrict__ w1t, __bf16* __restrict__ w2t,
                             __bf16* __restrict__ wihb, __bf16* __restrict__ whhb,
                             float* __restrict__ bsum)
{
    int idx = blockIdx.x * blockDim.x + threadIdx.x;
    if (idx < 16384) {                       // W1^T
        int j = idx >> 7, k = idx & 127;
        w1t[idx] = (__bf16)W1[k * 128 + j];
    } else if (idx < 32768) {                // W2^T
        int t = idx - 16384;
        int j = t >> 7, k = t & 127;
        w2t[t] = (__bf16)W2[k * 128 + j];
    } else if (idx < 32768 + 65536) {        // Wih
        int t = idx - 32768;
        wihb[t] = (__bf16)Wih[t];
    } else if (idx < 32768 + 131072) {       // Whh
        int t = idx - (32768 + 65536);
        whhb[t] = (__bf16)Whh[t];
    } else if (idx < 32768 + 131072 + 512) { // bias sum
        int t = idx - (32768 + 131072);
        bsum[t] = bih[t] + bhh[t];
    }
}

__global__ void zero_hc(float* __restrict__ h, float* __restrict__ c)
{
    int idx = blockIdx.x * blockDim.x + threadIdx.x;
    if (idx < NH) { h[idx] = 0.0f; c[idx] = 0.0f; }
}

// ---------------------------------------------------------------------------
// Degree / norm (deg counts in-edges on dest plus self-loop, per reference)
// ---------------------------------------------------------------------------
__global__ void deg_init(float* __restrict__ deg)
{
    int i = blockIdx.x * blockDim.x + threadIdx.x;
    if (i < NNODES) deg[i] = 1.0f;           // self loop
}

__global__ void deg_edges(const int* __restrict__ row, float* __restrict__ deg)
{
    int e = blockIdx.x * blockDim.x + threadIdx.x;
    if (e < NEDGES) atomicAdd(&deg[row[e]], 1.0f);
}

__global__ void dinv_kernel(const float* __restrict__ deg, float* __restrict__ dinv)
{
    int i = blockIdx.x * blockDim.x + threadIdx.x;
    if (i < NNODES) {
        float d = deg[i];
        dinv[i] = (d > 0.0f) ? rsqrtf(d) : 0.0f;
    }
}

// out[i,:] = h[i,:] * dinv[i]^2   (self-loop contribution)
__global__ void self_init(const float* __restrict__ h, const float* __restrict__ dinv,
                          float* __restrict__ out)
{
    int idx = blockIdx.x * blockDim.x + threadIdx.x;
    if (idx >= NH) return;
    int n = idx >> 7;
    float w = dinv[n];
    out[idx] = h[idx] * (w * w);
}

// 32 lanes per edge, float4 each: gather h[c,:] (512B coalesced), atomic-add
// into out[r,:]. out (25.6MB) lives in L2, atomics resolve there.
__global__ void agg_edges(const int* __restrict__ row, const int* __restrict__ col,
                          const float* __restrict__ dinv, const float* __restrict__ h,
                          float* __restrict__ out)
{
    unsigned gid = blockIdx.x * blockDim.x + threadIdx.x;
    int e    = gid >> 5;
    int lane = gid & 31;
    if (e >= NEDGES) return;
    int r = row[e], c = col[e];
    float w = dinv[r] * dinv[c];
    const float4* hv = (const float4*)(h + (size_t)c * HDIM);
    float*        op = out + (size_t)r * HDIM + lane * 4;
    float4 v = hv[lane];
    atomicAdd(op + 0, v.x * w);
    atomicAdd(op + 1, v.y * w);
    atomicAdd(op + 2, v.z * w);
    atomicAdd(op + 3, v.w * w);
}

__global__ void bias_relu(const float* __restrict__ in, const float* __restrict__ b,
                          float* __restrict__ out)
{
    int idx = blockIdx.x * blockDim.x + threadIdx.x;
    if (idx >= NH) return;
    out[idx] = fmaxf(in[idx] + b[idx & 127], 0.0f);
}

// ---------------------------------------------------------------------------
// WMMA GEMM:  C[M,Ncols] = A1[M,K] * B1t^T (+ A2[M,K] * B2t^T) (+ bias[col])
// A: f32 row-major (converted to bf16 in registers).
// B?t: bf16, row j = output column j, K contiguous.
// Wave tile 16x64 (4 x v8f acc); block = 8 waves = 128 rows x 64 cols.
// Fragment layouts per CDNA5 ISA 7.12.2 (16-bit A 16x32, B 32x16, f32 C 16x16).
// ---------------------------------------------------------------------------
__global__ __launch_bounds__(256)
void gemm_wmma_bf16(const float* __restrict__ A1, const __bf16* __restrict__ B1,
                    const float* __restrict__ A2, const __bf16* __restrict__ B2,
                    const float* __restrict__ bias, float* __restrict__ C,
                    int M, int Ncols, int K)
{
    const int wave = threadIdx.x >> 5;
    const int lane = threadIdx.x & 31;
    const int rowBase = blockIdx.y * 128 + wave * 16;
    if (rowBase >= M) return;                 // whole-wave guard, EXEC stays all-1
    const int colBase = blockIdx.x * 64;
    const int m  = lane & 15;                 // A row / B col / C col within tile
    const int hi = lane >> 4;                 // half-wave selector

    v8f acc[4];
#pragma unroll
    for (int tc = 0; tc < 4; ++tc) {
        float bv = bias ? bias[colBase + tc * 16 + m] : 0.0f;
#pragma unroll
        for (int r = 0; r < 8; ++r) acc[tc][r] = bv;
    }

#pragma unroll
    for (int pass = 0; pass < 2; ++pass) {
        const float*  A  = pass ? A2 : A1;
        const __bf16* Bt = pass ? B2 : B1;
        if (A == nullptr) continue;
        for (int kb = 0; kb < K; kb += 32) {
            // ---- A fragment: lane m holds M=m; halves 0..7 -> K=kb+hi*8+i,
            //      halves 8..15 -> K=kb+16+hi*8+i
            const float* ap = A + (size_t)(rowBase + m) * K + kb + hi * 8;
            const float4* ap4 = (const float4*)ap;
            float4 x0 = ap4[0], x1 = ap4[1];   // K = base .. base+7
            float4 y0 = ap4[4], y1 = ap4[5];   // K = base+16 .. base+23
            float av[16] = { x0.x, x0.y, x0.z, x0.w, x1.x, x1.y, x1.z, x1.w,
                             y0.x, y0.y, y0.z, y0.w, y1.x, y1.y, y1.z, y1.w };
            v16bf a;
#pragma unroll
            for (int i = 0; i < 16; ++i) a[i] = (__bf16)av[i];
            // ---- 4 column tiles: B fragment is two 8-bf16 (16B) runs, K=kb+hi*8+i and kb+16+hi*8+i
#pragma unroll
            for (int tc = 0; tc < 4; ++tc) {
                const __bf16* bp = Bt + (size_t)(colBase + tc * 16 + m) * K + kb + hi * 8;
                typedef __attribute__((ext_vector_type(8))) __bf16 v8bf_;
                v8bf_ b0 = *(const v8bf_*)bp, b1 = *(const v8bf_*)(bp + 16);
                v16bf b = __builtin_shufflevector(b0, b1, 0, 1, 2, 3, 4, 5, 6, 7, 8, 9, 10, 11, 12, 13, 14, 15);
                acc[tc] = __builtin_amdgcn_wmma_f32_16x16x32_bf16(
                    false, a, false, b, (short)0, acc[tc], false, false);
            }
        }
    }

    // ---- store: C VGPR r -> row rowBase + hi*8 + r, col = colBase+tc*16+m
#pragma unroll
    for (int tc = 0; tc < 4; ++tc) {
        const int col = colBase + tc * 16 + m;
        float* cp = C + (size_t)(rowBase + hi * 8) * Ncols + col;
#pragma unroll
        for (int r = 0; r < 8; ++r) cp[(size_t)r * Ncols] = acc[tc][r];
    }
}

// ---------------------------------------------------------------------------
// LSTM pointwise: gates [N,512] order i,f,g,o; update c, h
// ---------------------------------------------------------------------------
__device__ __forceinline__ float sigmoidf_(float x) { return 1.0f / (1.0f + __expf(-x)); }

__global__ void lstm_pointwise(const float* __restrict__ gates, float* __restrict__ cst,
                               float* __restrict__ h)
{
    int idx = blockIdx.x * blockDim.x + threadIdx.x;
    if (idx >= NH) return;
    int n = idx >> 7, j = idx & 127;
    const float* g = gates + (size_t)n * 512;
    float i  = sigmoidf_(g[j]);
    float f  = sigmoidf_(g[128 + j]);
    float gg = tanhf(g[256 + j]);
    float o  = sigmoidf_(g[384 + j]);
    float c  = f * cst[idx] + i * gg;
    cst[idx] = c;
    h[idx]   = o * tanhf(c);
}

// ---------------------------------------------------------------------------
extern "C" void kernel_launch(void* const* d_in, const int* in_sizes, int n_in,
                              void* d_out, int out_size, void* d_ws, size_t ws_size,
                              hipStream_t stream)
{
    const float* node_feats = (const float*)d_in[0];  // [T,N,128]
    const int*   edges      = (const int*)  d_in[1];  // [T,2,E]
    const float* W1  = (const float*)d_in[2];
    const float* b1  = (const float*)d_in[3];
    const float* W2  = (const float*)d_in[4];
    const float* b2  = (const float*)d_in[5];
    const float* Wih = (const float*)d_in[6];
    const float* Whh = (const float*)d_in[7];
    const float* bih = (const float*)d_in[8];
    const float* bhh = (const float*)d_in[9];
    float* hout = (float*)d_out;                      // LSTM hidden state lives here

    // workspace carve-up
    char*  ws  = (char*)d_ws;
    size_t off = 0;
    auto carve = [&](size_t bytes) -> void* {
        void* p = ws + off;
        off += (bytes + 255) & ~(size_t)255;
        return p;
    };
    __bf16* w1t  = (__bf16*)carve(16384  * sizeof(__bf16));
    __bf16* w2t  = (__bf16*)carve(16384  * sizeof(__bf16));
    __bf16* wihb = (__bf16*)carve(65536  * sizeof(__bf16));
    __bf16* whhb = (__bf16*)carve(65536  * sizeof(__bf16));
    float*  bsum = (float*) carve(512    * sizeof(float));
    float*  deg  = (float*) carve(NNODES * sizeof(float));
    float*  dinv = (float*) carve(NNODES * sizeof(float));
    float*  bufA = (float*) carve((size_t)NH * sizeof(float));
    float*  bufB = (float*) carve((size_t)NH * sizeof(float));
    float*  cst  = (float*) carve((size_t)NH * sizeof(float));
    float*  gates= (float*) carve((size_t)NNODES * 512 * sizeof(float));
    (void)ws_size; (void)in_sizes; (void)n_in; (void)out_size;

    const int TPB = 256;
    const int nhBlocks   = (NH + TPB - 1) / TPB;
    const int nodeBlocks = (NNODES + TPB - 1) / TPB;
    const int edgeBlocks = (NEDGES + TPB - 1) / TPB;
    const int aggBlocks  = (int)(((size_t)NEDGES * 32 + TPB - 1) / TPB);
    const dim3 gemmGrid128(128 / 64, (NNODES + 127) / 128);  // Ncols=128
    const dim3 gemmGrid512(512 / 64, (NNODES + 127) / 128);  // Ncols=512

    prep_weights<<<(32768 + 131072 + 512 + TPB - 1) / TPB, TPB, 0, stream>>>(
        W1, W2, Wih, Whh, bih, bhh, w1t, w2t, wihb, whhb, bsum);
    zero_hc<<<nhBlocks, TPB, 0, stream>>>(hout, cst);

    for (int t = 0; t < TSTEPS; ++t) {
        const float* Xt  = node_feats + (size_t)t * NH;
        const int*   row = edges + (size_t)t * 2 * NEDGES;
        const int*   col = row + NEDGES;

        // symmetric norm
        deg_init<<<nodeBlocks, TPB, 0, stream>>>(deg);
        deg_edges<<<edgeBlocks, TPB, 0, stream>>>(row, deg);
        dinv_kernel<<<nodeBlocks, TPB, 0, stream>>>(deg, dinv);

        // GCN layer 1
        gemm_wmma_bf16<<<gemmGrid128, TPB, 0, stream>>>(
            Xt, w1t, nullptr, nullptr, nullptr, bufA, NNODES, 128, 128);
        self_init<<<nhBlocks, TPB, 0, stream>>>(bufA, dinv, bufB);
        agg_edges<<<aggBlocks, TPB, 0, stream>>>(row, col, dinv, bufA, bufB);
        bias_relu<<<nhBlocks, TPB, 0, stream>>>(bufB, b1, bufA);

        // GCN layer 2
        gemm_wmma_bf16<<<gemmGrid128, TPB, 0, stream>>>(
            bufA, w2t, nullptr, nullptr, nullptr, bufB, NNODES, 128, 128);
        self_init<<<nhBlocks, TPB, 0, stream>>>(bufB, dinv, bufA);
        agg_edges<<<aggBlocks, TPB, 0, stream>>>(row, col, dinv, bufB, bufA);
        bias_relu<<<nhBlocks, TPB, 0, stream>>>(bufA, b2, bufB);   // bufB = emb_t

        // LSTM gates: emb_t @ Wih^T + h @ Whh^T + (bih+bhh), dual-K WMMA GEMM
        gemm_wmma_bf16<<<gemmGrid512, TPB, 0, stream>>>(
            bufB, wihb, hout, whhb, bsum, gates, NNODES, 512, 128);
        lstm_pointwise<<<nhBlocks, TPB, 0, stream>>>(gates, cst, hout);
    }
}
